// GmmFull_76751065580080
// MI455X (gfx1250) — matrix operand order repaired and checked by reading, not compiled, not run on hardware
//
#include <hip/hip_runtime.h>

#define NROWS 262144
#define FDIM  128
#define DDIM  64
#define KCOMP 16
#define MBLK  64                  // rows per block
#define XPAD32 132                // padded f32 row stride (floats): 528 B/row
#define XPAD16 136                // padded f16 row stride (halves): 272 B/row, 4-bank skew

typedef _Float16 v16h __attribute__((ext_vector_type(16)));
typedef _Float16 v8h  __attribute__((ext_vector_type(8)));
typedef _Float16 v4h  __attribute__((ext_vector_type(4)));
typedef float    v8f  __attribute__((ext_vector_type(8)));
typedef float    v4f  __attribute__((ext_vector_type(4)));

// --- CDNA5 async global->LDS copy (ASYNCcnt-tracked DMA, no VGPR pass-through)
__device__ __forceinline__ void async_copy_b128(unsigned lds_off, const void* gaddr) {
    asm volatile("global_load_async_to_lds_b128 %0, %1, off"
                 :: "v"(lds_off), "v"(gaddr) : "memory");
}
__device__ __forceinline__ void wait_asynccnt0() {
    asm volatile("s_wait_asynccnt 0" ::: "memory");
}

// ---------------------------------------------------------------------------
// Prep: per-component Cholesky, triangular inverse, fold into embedding.
//   C_k = L_k^{-1} @ W_embed   [D, F]  (f16)
//   m_k = L_k^{-1} @ mu_k      [D]     (f32)
//   cst_k = -0.5*D*log(2pi) - sum(log diag L_k) + log_softmax(logits)_k
// ---------------------------------------------------------------------------
__global__ void prep_kernel(const float* __restrict__ W,       // [D,F]
                            const float* __restrict__ logits,  // [K]
                            const float* __restrict__ mus,     // [K,D]
                            const float* __restrict__ sigmas,  // [K,D,D]
                            _Float16* __restrict__ Cmat,       // [K,D,F]
                            float* __restrict__ mvec,          // [K,D]
                            float* __restrict__ cst)           // [K]
{
    __shared__ float Ls[DDIM][DDIM + 1];
    __shared__ float As[DDIM][DDIM + 1];
    const int k = blockIdx.x;
    const int t = threadIdx.x;  // 0..63

    for (int j = 0; j < DDIM; ++j)
        Ls[t][j] = sigmas[(size_t)k * DDIM * DDIM + (size_t)t * DDIM + j];
    __syncthreads();

    for (int j = 0; j < DDIM; ++j) {
        if (t == j) {
            float s = Ls[j][j];
            for (int p = 0; p < j; ++p) s -= Ls[j][p] * Ls[j][p];
            Ls[j][j] = sqrtf(s);
        }
        __syncthreads();
        if (t > j) {
            float s = Ls[t][j];
            for (int p = 0; p < j; ++p) s -= Ls[t][p] * Ls[j][p];
            Ls[t][j] = s / Ls[j][j];
        }
        __syncthreads();
    }

    // A = L^{-1}: thread t forward-solves L a = e_t (columns independent).
    {
        const int c = t;
        for (int i = 0; i < DDIM; ++i) {
            if (i < c) { As[i][c] = 0.0f; }
            else {
                float s = (i == c) ? 1.0f : 0.0f;
                for (int j = c; j < i; ++j) s -= Ls[i][j] * As[j][c];
                As[i][c] = s / Ls[i][i];
            }
        }
    }
    __syncthreads();

    for (int f = 0; f < FDIM; ++f) {
        float s = 0.0f;
        for (int j = 0; j <= t; ++j) s += As[t][j] * W[(size_t)j * FDIM + f];
        Cmat[((size_t)k * DDIM + t) * FDIM + f] = (_Float16)s;
    }
    {
        float s = 0.0f;
        for (int j = 0; j <= t; ++j) s += As[t][j] * mus[k * DDIM + j];
        mvec[k * DDIM + t] = s;
    }
    if (t == 0) {
        float hld = 0.0f;
        for (int d = 0; d < DDIM; ++d) hld += logf(Ls[d][d]);
        float mx = logits[0];
        for (int i = 1; i < KCOMP; ++i) mx = fmaxf(mx, logits[i]);
        float se = 0.0f;
        for (int i = 0; i < KCOMP; ++i) se += expf(logits[i] - mx);
        const float logmix = logits[k] - (mx + logf(se));
        cst[k] = -0.5f * (float)DDIM * 1.8378770664093453f - hld + logmix;
    }
}

// ---------------------------------------------------------------------------
// Fused kernel: Y = x @ C^T via WMMA, then maha / logsumexp epilogue.
// Block = 256 threads (8 waves), 64 rows x all K*D=1024 columns.
// Wave w owns components {2w, 2w+1}: 4 row subtiles x 8 col tiles = 32 accs.
// x tile staged fp32 -> LDS by async DMA, converted ONCE to a bank-padded
// f16 tile in the same LDS buffer; the WMMA loop is then pure ds/vmem/wmma.
// f-loop kept ROLLED: one f-step of B fragments live -> no register spills.
// ---------------------------------------------------------------------------
__global__ void __launch_bounds__(256)
gmm_kernel(const float* __restrict__ x,        // [N,F]
           const _Float16* __restrict__ Cmat,  // [K,D,F]
           const float* __restrict__ mvec,     // [K,D]
           const float* __restrict__ cst,      // [K]
           float* __restrict__ partial)        // [gridDim.x]
{
    // Union buffer: fp32 staging view [64][132], then f16 view [64][136].
    __shared__ __align__(16) char xbuf[MBLK * XPAD32 * 4];   // 33.8 KB
    __shared__ float clp[KCOMP][MBLK];                       // 4 KB
    __shared__ float lpred[MBLK];

    float*    xs32 = (float*)xbuf;
    _Float16* xs16 = (_Float16*)xbuf;

    const int tid  = threadIdx.x;
    const int lane = tid & 31;
    const int wave = tid >> 5;
    const int row0 = blockIdx.x * MBLK;

    // ---- async DMA: 64x128 fp32 tile -> LDS (16 B per lane-op, 8 ops/thread)
    {
        const unsigned ldsbase = (unsigned)(unsigned long long)(const void*)xs32;
#pragma unroll
        for (int i = 0; i < 8; ++i) {
            const int q  = tid + i * 256;   // 0..2047 segments of 16 B
            const int r  = q >> 5;          // row 0..63
            const int cs = q & 31;          // 16 B segment within row
            const float* g = x + (size_t)(row0 + r) * FDIM + cs * 4;
            const unsigned loff = ldsbase + (unsigned)(r * XPAD32 + cs * 4) * 4u;
            async_copy_b128(loff, (const void*)g);
        }
    }
    wait_asynccnt0();
    __syncthreads();

    // ---- one-pass convert in place: pull fp32 to regs, barrier, store f16.
    v4f stg[8];
#pragma unroll
    for (int i = 0; i < 8; ++i) {
        const int q  = tid + i * 256;
        const int r  = q >> 5;
        const int cs = q & 31;
        stg[i] = *(const v4f*)&xs32[r * XPAD32 + cs * 4];
    }
    __syncthreads();   // all reads done before any f16 overwrite
#pragma unroll
    for (int i = 0; i < 8; ++i) {
        const int q  = tid + i * 256;
        const int r  = q >> 5;
        const int cs = q & 31;
        v4h h;
#pragma unroll
        for (int e = 0; e < 4; ++e) h[e] = (_Float16)stg[i][e];
        *(v4h*)(xs16 + r * XPAD16 + cs * 4) = h;
    }
    __syncthreads();

    const int hi = lane >> 4;   // half-wave select per ISA 16-bit A/B layouts
    const int ln = lane & 15;
    const int kbase = wave * 2;

    const v8f zv = {0.f, 0.f, 0.f, 0.f, 0.f, 0.f, 0.f, 0.f};
    v8f acc[4][8];
#pragma unroll
    for (int s = 0; s < 4; ++s)
#pragma unroll
        for (int t2 = 0; t2 < 8; ++t2) acc[s][t2] = zv;

    // K-dim loop over features, 32 per WMMA step. Keep ROLLED (reg pressure).
#pragma clang loop unroll(disable)
    for (int f0 = 0; f0 < FDIM; f0 += 32) {
        // Preload all 8 B fragments of this f-step (one clause, 4 WMMAs each).
        v16h b[8];
#pragma unroll
        for (int j = 0; j < 8; ++j) {
            const int c  = j >> 2;
            const int dt = j & 3;
            const int d  = dt * 16 + ln;                    // output column index
            b[j] = *(const v16h*)(Cmat + ((size_t)(kbase + c) * DDIM + d) * FDIM
                                  + f0 + hi * 16);
        }
#pragma unroll
        for (int s = 0; s < 4; ++s) {
            // A fragment (16x32 f16): two 16 B LDS loads, no VALU.
            // lane<16 row=ln holds K {f0..+15}; lane>=16 {+16..+31}.
            const int m = s * 16 + ln;
            const _Float16* bp = xs16 + m * XPAD16 + f0 + hi * 16;
            const v8h lo = *(const v8h*)(bp);
            const v8h hh = *(const v8h*)(bp + 8);
            const v16h a = __builtin_shufflevector(lo, hh, 0, 1, 2, 3, 4, 5, 6, 7,
                                                   8, 9, 10, 11, 12, 13, 14, 15);
#pragma unroll
            for (int j = 0; j < 8; ++j) {
                acc[s][j] = __builtin_amdgcn_wmma_f32_16x16x32_f16(
                    false, a, false, b[j], (short)0, acc[s][j], false, false);
            }
        }
    }

    // Epilogue: maha_nk = sum_d (y - m)^2, comp_lp into LDS.
#pragma unroll
    for (int c = 0; c < 2; ++c) {
        const int k = kbase + c;
        float pm[4][8];
#pragma unroll
        for (int s = 0; s < 4; ++s)
#pragma unroll
            for (int r = 0; r < 8; ++r) pm[s][r] = 0.0f;

#pragma unroll
        for (int dt = 0; dt < 4; ++dt) {
            const int d = dt * 16 + ln;
            const float mkd = mvec[k * DDIM + d];
#pragma unroll
            for (int s = 0; s < 4; ++s)
#pragma unroll
                for (int r = 0; r < 8; ++r) {
                    const float v = acc[s][c * 4 + dt][r] - mkd;
                    pm[s][r] += v * v;
                }
        }
        // Sum over the 16 lanes holding distinct d values (per half-wave).
#pragma unroll
        for (int mm = 1; mm < 16; mm <<= 1)
#pragma unroll
            for (int s = 0; s < 4; ++s)
#pragma unroll
                for (int r = 0; r < 8; ++r)
                    pm[s][r] += __shfl_xor(pm[s][r], mm, 16);

        if (ln == 0) {
            const float ck = cst[k];
#pragma unroll
            for (int s = 0; s < 4; ++s)
#pragma unroll
                for (int r = 0; r < 8; ++r)
                    clp[k][s * 16 + r + 8 * hi] = ck - 0.5f * pm[s][r];
        }
    }
    __syncthreads();

    // Per-row logsumexp over K, then deterministic block sum.
    if (tid < MBLK) {
        const int m = tid;
        float mx = clp[0][m];
#pragma unroll
        for (int k = 1; k < KCOMP; ++k) mx = fmaxf(mx, clp[k][m]);
        float se = 0.0f;
#pragma unroll
        for (int k = 0; k < KCOMP; ++k) se += __expf(clp[k][m] - mx);
        lpred[m] = mx + __logf(se);
    }
    __syncthreads();
    if (tid == 0) {
        float s = 0.0f;
#pragma unroll
        for (int m = 0; m < MBLK; ++m) s += lpred[m];
        partial[blockIdx.x] = s;
    }
}

// Deterministic final reduction: out = -mean(lp).
__global__ void finalize_kernel(const float* __restrict__ partial, int nblk,
                                float* __restrict__ out) {
    __shared__ float red[256];
    float s = 0.0f;
    for (int i = threadIdx.x; i < nblk; i += 256) s += partial[i];
    red[threadIdx.x] = s;
    __syncthreads();
    for (int off = 128; off > 0; off >>= 1) {
        if ((int)threadIdx.x < off) red[threadIdx.x] += red[threadIdx.x + off];
        __syncthreads();
    }
    if (threadIdx.x == 0) out[0] = -red[0] / (float)NROWS;
}

extern "C" void kernel_launch(void* const* d_in, const int* in_sizes, int n_in,
                              void* d_out, int out_size, void* d_ws, size_t ws_size,
                              hipStream_t stream) {
    (void)in_sizes; (void)n_in; (void)out_size; (void)ws_size;
    const float* x      = (const float*)d_in[0];
    const float* W      = (const float*)d_in[1];
    const float* logits = (const float*)d_in[2];
    const float* mus    = (const float*)d_in[3];
    const float* sigmas = (const float*)d_in[4];
    float* out = (float*)d_out;

    char* ws = (char*)d_ws;
    _Float16* Cmat = (_Float16*)ws;                              // 262144 B
    float* mvec    = (float*)(ws + 262144);                      //   4096 B
    float* cst     = (float*)(ws + 262144 + 4096);               //     64 B
    float* partial = (float*)(ws + 262144 + 4096 + 64);          //  16384 B

    const int nblk = NROWS / MBLK;  // 4096
    prep_kernel<<<KCOMP, 64, 0, stream>>>(W, logits, mus, sigmas, Cmat, mvec, cst);
    gmm_kernel<<<nblk, 256, 0, stream>>>(x, Cmat, mvec, cst, partial);
    finalize_kernel<<<1, 256, 0, stream>>>(partial, nblk, out);
}
